// MOG_LSTM_31980326486669
// MI455X (gfx1250) — compile-verified, matrix operand order checked by reading, not run on hardware
//
#include <hip/hip_runtime.h>
#include <hip/hip_bf16.h>

#define D_ 1024
#define H_ 1024
#define V_ 32000
#define B_ 32
#define T_ 64
#define NG_ 4096   // 4*H

typedef __attribute__((ext_vector_type(16))) __bf16 v16bf;
typedef __attribute__((ext_vector_type(8)))  __bf16 v8bf;
typedef __attribute__((ext_vector_type(8)))  float  v8f;

__device__ __forceinline__ __bf16 to_bf16(float f) {
  unsigned int u = __builtin_bit_cast(unsigned int, f);
  unsigned int r = (u + 0x7FFFu + ((u >> 16) & 1u)) >> 16;
  return __builtin_bit_cast(__bf16, (unsigned short)r);
}

__device__ __forceinline__ float sigmoidf(float x) {
  return 1.0f / (1.0f + __expf(-x));
}

// Load one 16x32 bf16 WMMA operand fragment for this lane.
// Per CDNA5 ISA layout (16-bit A 16x32): lane group g holds
// elements e<8 at K = kb+8g+e and e>=8 at K = kb+16+8g+(e-8).
// Caller passes p = &row_base[kb + 8*g]; we read 8 elems at p and 8 at p+16.
__device__ __forceinline__ v16bf load_frag(const __bf16* p) {
  v8bf lo = *(const v8bf*)(p);
  v8bf hi = *(const v8bf*)(p + 16);
  return __builtin_shufflevector(lo, hi, 0,1,2,3,4,5,6,7,8,9,10,11,12,13,14,15);
}

__device__ __forceinline__ v8f wmma_bf16(v16bf a, v16bf b, v8f c) {
  return __builtin_amdgcn_wmma_f32_16x16x32_bf16(false, a, false, b, (short)0, c, false, false);
}

// ---------------- utility kernels ----------------

__global__ void k_f32_to_bf16(const float* __restrict__ src, __bf16* __restrict__ dst, long n) {
  long i = (long)blockIdx.x * blockDim.x + threadIdx.x;
  long stride = (long)gridDim.x * blockDim.x;
  for (; i < n; i += stride) dst[i] = to_bf16(src[i]);
}

__global__ void k_zero(float* __restrict__ p, int n) {
  int i = blockIdx.x * blockDim.x + threadIdx.x;
  if (i < n) p[i] = 0.0f;
}

// Gather embeddings: one block per (b,t) row; copies 1024 floats.
__global__ void k_gather(const int* __restrict__ seq, const float* __restrict__ emb,
                         float* __restrict__ xs_all) {
  int row = blockIdx.x;                 // row = b*T + t
  int tok = seq[row];
  const float4* s = (const float4*)(emb + (size_t)tok * D_);
  float4* d = (float4*)(xs_all + (size_t)row * D_);
  d[threadIdx.x] = s[threadIdx.x];
}

// ---------------- mogrify step: tgt_out = 2*sigmoid(src @ W^T + bias) * tgt_in ----------------
// src: (32,1024) f32. tgt_in: rows at tgt_in + m*tgt_stride (lets step 0 read strided xs_all
// or h1 directly, fusing away the init/copy kernels). Wb: (1024,1024) bf16 row-major.
// Grid: 8 blocks x 256 threads; each wave computes a 32x16 tile of the matmul.
__global__ void k_mogrify(float* __restrict__ tgt_out, const float* __restrict__ tgt_in,
                          long tgt_stride, const float* __restrict__ src,
                          const __bf16* __restrict__ Wb, const float* __restrict__ bias) {
  __shared__ __bf16 Sb[B_ * H_];   // 64 KB bf16 copy of src
  int tid = threadIdx.x;
  for (int i = tid; i < B_ * H_; i += 256) Sb[i] = to_bf16(src[i]);
  __syncthreads();

  int wave = tid >> 5, lane = tid & 31;
  int g = lane >> 4, ln = lane & 15;
  int n0 = (blockIdx.x * 8 + wave) * 16;

  v8f c0 = {}, c1 = {};
  #pragma unroll 2
  for (int kb = 0; kb < H_; kb += 32) {
    v16bf a0 = load_frag(&Sb[(size_t)ln * H_ + kb + 8 * g]);
    v16bf a1 = load_frag(&Sb[(size_t)(16 + ln) * H_ + kb + 8 * g]);
    v16bf bb = load_frag(&Wb[(size_t)(n0 + ln) * H_ + kb + 8 * g]);
    c0 = wmma_bf16(a0, bb, c0);
    c1 = wmma_bf16(a1, bb, c1);
  }

  int col = n0 + ln;
  float bv = bias[col];
  #pragma unroll
  for (int r = 0; r < 8; ++r) {
    int m0 = r + 8 * g;
    int m1 = 16 + r + 8 * g;
    float t0 = tgt_in[(size_t)m0 * tgt_stride + col];
    float t1 = tgt_in[(size_t)m1 * tgt_stride + col];
    tgt_out[(size_t)m0 * H_ + col] = t0 * 2.0f * sigmoidf(c0[r] + bv);
    tgt_out[(size_t)m1 * H_ + col] = t1 * 2.0f * sigmoidf(c1[r] + bv);
  }
}

// ---------------- LSTM gate GEMM: g = x@W_ih^T + b_ih + h@W_hh^T + b_hh ----------------
// Two phases sharing one 64KB LDS buffer. Grid: 32 blocks x 256 threads (N = 4096).
__global__ void k_gates(const float* __restrict__ x, const float* __restrict__ h,
                        const __bf16* __restrict__ Wih, const __bf16* __restrict__ Whh,
                        const float* __restrict__ bih, const float* __restrict__ bhh,
                        float* __restrict__ gbuf) {
  __shared__ __bf16 Sb[B_ * H_];
  int tid = threadIdx.x;
  int wave = tid >> 5, lane = tid & 31;
  int g = lane >> 4, ln = lane & 15;
  int n0 = (blockIdx.x * 8 + wave) * 16;

  v8f c0 = {}, c1 = {};

  // phase 1: x @ W_ih^T
  for (int i = tid; i < B_ * H_; i += 256) Sb[i] = to_bf16(x[i]);
  __syncthreads();
  #pragma unroll 2
  for (int kb = 0; kb < H_; kb += 32) {
    v16bf a0 = load_frag(&Sb[(size_t)ln * H_ + kb + 8 * g]);
    v16bf a1 = load_frag(&Sb[(size_t)(16 + ln) * H_ + kb + 8 * g]);
    v16bf bb = load_frag(&Wih[(size_t)(n0 + ln) * H_ + kb + 8 * g]);
    c0 = wmma_bf16(a0, bb, c0);
    c1 = wmma_bf16(a1, bb, c1);
  }
  __syncthreads();

  // phase 2: + h @ W_hh^T
  for (int i = tid; i < B_ * H_; i += 256) Sb[i] = to_bf16(h[i]);
  __syncthreads();
  #pragma unroll 2
  for (int kb = 0; kb < H_; kb += 32) {
    v16bf a0 = load_frag(&Sb[(size_t)ln * H_ + kb + 8 * g]);
    v16bf a1 = load_frag(&Sb[(size_t)(16 + ln) * H_ + kb + 8 * g]);
    v16bf bb = load_frag(&Whh[(size_t)(n0 + ln) * H_ + kb + 8 * g]);
    c0 = wmma_bf16(a0, bb, c0);
    c1 = wmma_bf16(a1, bb, c1);
  }

  int col = n0 + ln;
  float bv = bih[col] + bhh[col];
  #pragma unroll
  for (int r = 0; r < 8; ++r) {
    gbuf[(size_t)(r + 8 * g) * NG_ + col]      = c0[r] + bv;
    gbuf[(size_t)(16 + r + 8 * g) * NG_ + col] = c1[r] + bv;
  }
}

// ---------------- LSTM cell elementwise ----------------
__global__ void k_cell(const float* __restrict__ gbuf, float* __restrict__ c,
                       float* __restrict__ h, float* __restrict__ hs_out,
                       __bf16* __restrict__ hsb, int t) {
  int idx = blockIdx.x * blockDim.x + threadIdx.x;   // 32768
  int b = idx >> 10, n = idx & 1023;
  const float* gr = gbuf + (size_t)b * NG_;
  float iv = sigmoidf(gr[n]);
  float fv = sigmoidf(gr[H_ + n]);
  float gv = tanhf(gr[2 * H_ + n]);
  float ov = sigmoidf(gr[3 * H_ + n]);
  float cv = fv * c[idx] + iv * gv;
  c[idx] = cv;
  float hv = ov * tanhf(cv);
  h[idx] = hv;
  if (hs_out) {
    size_t row = (size_t)b * T_ + t;
    hs_out[row * H_ + n] = hv;
    hsb[row * H_ + n] = to_bf16(hv);
  }
}

// ---------------- Output projection: out = hs @ emb^T + fc_b ----------------
// M=2048, N=32000, K=1024. Block = 256 thr (8 waves) computes 64x128 tile.
// B tiles (128 cols x 32 K of bf16 = 8KB) are staged into LDS with
// GLOBAL_LOAD_ASYNC_TO_LDS_B128, double-buffered against the WMMA compute
// (ASYNCcnt pipeline). A comes straight from global (4MB total, L2-resident).
__global__ void k_proj(const __bf16* __restrict__ hsb, const __bf16* __restrict__ embb,
                       const float* __restrict__ fcb, float* __restrict__ out) {
  __shared__ __bf16 Bt[2][128 * 32];   // two 8KB B-tile buffers

  int tid = threadIdx.x;
  int wave = tid >> 5, lane = tid & 31;
  int g = lane >> 4, ln = lane & 15;
  int wm = wave >> 2, wn = wave & 3;
  int mbase = blockIdx.y * 64 + wm * 32;
  int nblock = blockIdx.x * 128;

  // This thread's two async-load slots per stage (512 x 16B loads per 8KB tile).
  int l0 = tid * 2, l1 = tid * 2 + 1;
  int r0 = l0 >> 2, q0 = l0 & 3;     // row 0..127, 16B-quarter 0..3
  int r1 = l1 >> 2, q1 = l1 & 3;
  const __bf16* gb0 = embb + (size_t)(nblock + r0) * H_ + q0 * 8;
  const __bf16* gb1 = embb + (size_t)(nblock + r1) * H_ + q1 * 8;
  // Generic->LDS: addr[31:0] of a generic shared pointer is the LDS byte offset.
  unsigned ldsBase = (unsigned)(size_t)&Bt[0][0];
  unsigned o0 = (unsigned)(r0 * 64 + q0 * 16);
  unsigned o1 = (unsigned)(r1 * 64 + q1 * 16);

  auto issue = [&](int buf, int kb) {
    unsigned d0 = ldsBase + (unsigned)buf * 8192u + o0;
    unsigned d1 = ldsBase + (unsigned)buf * 8192u + o1;
    unsigned long long a0 = (unsigned long long)(size_t)(gb0 + kb);
    unsigned long long a1 = (unsigned long long)(size_t)(gb1 + kb);
    asm volatile("global_load_async_to_lds_b128 %0, %1, off\n\t"
                 "global_load_async_to_lds_b128 %2, %3, off"
                 :
                 : "v"(d0), "v"(a0), "v"(d1), "v"(a1)
                 : "memory");
  };

  v8f a00 = {}, a01 = {}, a10 = {}, a11 = {};

  const int NSTEP = H_ / 32;   // 32 k-steps
  issue(0, 0);
  for (int s = 0; s < NSTEP; ++s) {
    int buf = s & 1;
    if (s + 1 < NSTEP) {
      issue(buf ^ 1, (s + 1) * 32);
      asm volatile("s_wait_asynccnt 0x2" ::: "memory");  // current stage landed
    } else {
      asm volatile("s_wait_asynccnt 0x0" ::: "memory");
    }
    __syncthreads();   // all waves' slices of this stage visible

    int kb = s * 32;
    const __bf16* ap = hsb + (size_t)(mbase + ln) * H_ + kb + 8 * g;
    __builtin_prefetch(ap + 32, 0, 1);
    v16bf fa0 = load_frag(ap);
    v16bf fa1 = load_frag(ap + 16 * H_);
    const __bf16* bp = &Bt[buf][(size_t)(wn * 32 + ln) * 32 + 8 * g];
    v16bf fb0 = load_frag(bp);
    v16bf fb1 = load_frag(bp + 16 * 32);
    a00 = wmma_bf16(fa0, fb0, a00);
    a01 = wmma_bf16(fa0, fb1, a01);
    a10 = wmma_bf16(fa1, fb0, a10);
    a11 = wmma_bf16(fa1, fb1, a11);

    __syncthreads();   // done reading buf before it is refilled two stages later
  }

  int col0 = nblock + wn * 32 + ln, col1 = col0 + 16;
  float b0 = fcb[col0], b1 = fcb[col1];
  #pragma unroll
  for (int r = 0; r < 8; ++r) {
    size_t row0 = (size_t)(mbase + r + 8 * g);
    size_t row1 = row0 + 16;
    out[row0 * V_ + col0] = a00[r] + b0;
    out[row0 * V_ + col1] = a01[r] + b1;
    out[row1 * V_ + col0] = a10[r] + b0;
    out[row1 * V_ + col1] = a11[r] + b1;
  }
}

// ---------------- host launcher ----------------

extern "C" void kernel_launch(void* const* d_in, const int* in_sizes, int n_in,
                              void* d_out, int out_size, void* d_ws, size_t ws_size,
                              hipStream_t stream) {
  const int*   seq    = (const int*)d_in[0];
  const float* emb    = (const float*)d_in[2];
  const float* mog1_W = (const float*)d_in[3];
  const float* mog1_b = (const float*)d_in[4];
  const float* W_ih1  = (const float*)d_in[5];
  const float* W_hh1  = (const float*)d_in[6];
  const float* b_ih1  = (const float*)d_in[7];
  const float* b_hh1  = (const float*)d_in[8];
  const float* mog2_W = (const float*)d_in[9];
  const float* mog2_b = (const float*)d_in[10];
  const float* W_ih2  = (const float*)d_in[11];
  const float* W_hh2  = (const float*)d_in[12];
  const float* b_ih2  = (const float*)d_in[13];
  const float* b_hh2  = (const float*)d_in[14];
  const float* fc_b   = (const float*)d_in[15];

  char* base = (char*)d_ws;
  size_t off = 0;
  auto alloc = [&](size_t bytes) -> void* {
    void* p = base + off;
    off += (bytes + 255) & ~(size_t)255;
    return p;
  };

  __bf16* emb_b  = (__bf16*)alloc((size_t)V_ * D_ * 2);
  __bf16* m1Wb   = (__bf16*)alloc((size_t)5 * H_ * H_ * 2);
  __bf16* m2Wb   = (__bf16*)alloc((size_t)5 * H_ * H_ * 2);
  __bf16* Wih1b  = (__bf16*)alloc((size_t)NG_ * D_ * 2);
  __bf16* Whh1b  = (__bf16*)alloc((size_t)NG_ * H_ * 2);
  __bf16* Wih2b  = (__bf16*)alloc((size_t)NG_ * H_ * 2);
  __bf16* Whh2b  = (__bf16*)alloc((size_t)NG_ * H_ * 2);
  __bf16* hsb    = (__bf16*)alloc((size_t)B_ * T_ * H_ * 2);
  float*  xs_all = (float*)alloc((size_t)B_ * T_ * D_ * 4);
  float*  x1     = (float*)alloc((size_t)B_ * H_ * 4);
  float*  h1     = (float*)alloc((size_t)B_ * H_ * 4);
  float*  c1     = (float*)alloc((size_t)B_ * H_ * 4);
  float*  x2     = (float*)alloc((size_t)B_ * H_ * 4);
  float*  h2     = (float*)alloc((size_t)B_ * H_ * 4);
  float*  c2     = (float*)alloc((size_t)B_ * H_ * 4);
  float*  gbuf   = (float*)alloc((size_t)B_ * NG_ * 4);

  // Weight / embedding conversion to bf16 (done every call; deterministic).
  k_f32_to_bf16<<<4096, 256, 0, stream>>>(emb,    emb_b, (long)V_ * D_);
  k_f32_to_bf16<<<2048, 256, 0, stream>>>(mog1_W, m1Wb,  (long)5 * H_ * H_);
  k_f32_to_bf16<<<2048, 256, 0, stream>>>(mog2_W, m2Wb,  (long)5 * H_ * H_);
  k_f32_to_bf16<<<2048, 256, 0, stream>>>(W_ih1,  Wih1b, (long)NG_ * D_);
  k_f32_to_bf16<<<2048, 256, 0, stream>>>(W_hh1,  Whh1b, (long)NG_ * H_);
  k_f32_to_bf16<<<2048, 256, 0, stream>>>(W_ih2,  Wih2b, (long)NG_ * H_);
  k_f32_to_bf16<<<2048, 256, 0, stream>>>(W_hh2,  Whh2b, (long)NG_ * H_);

  k_gather<<<B_ * T_, 256, 0, stream>>>(seq, emb, xs_all);

  const int SN = B_ * H_;          // 32768 state elements
  k_zero<<<SN / 256, 256, 0, stream>>>(h1, SN);
  k_zero<<<SN / 256, 256, 0, stream>>>(c1, SN);
  k_zero<<<SN / 256, 256, 0, stream>>>(h2, SN);
  k_zero<<<SN / 256, 256, 0, stream>>>(c2, SN);

  float* outF   = (float*)d_out;
  float* hs_out = outF + (size_t)B_ * T_ * V_;

  for (int t = 0; t < T_; ++t) {
    // Layer 1 mogrify. Step 0 reads x directly out of xs_all (strided) -> x1.
    for (int i = 0; i < 5; ++i) {
      const __bf16* Wb = m1Wb + (size_t)i * H_ * H_;
      const float*  bb = mog1_b + (size_t)i * H_;
      if (i == 0)
        k_mogrify<<<8, 256, 0, stream>>>(x1, xs_all + (size_t)t * D_, (long)T_ * D_, h1, Wb, bb);
      else if ((i & 1) == 0)
        k_mogrify<<<8, 256, 0, stream>>>(x1, x1, (long)H_, h1, Wb, bb);
      else
        k_mogrify<<<8, 256, 0, stream>>>(h1, h1, (long)H_, x1, Wb, bb);
    }
    k_gates<<<32, 256, 0, stream>>>(x1, h1, Wih1b, Whh1b, b_ih1, b_hh1, gbuf);
    k_cell<<<SN / 256, 256, 0, stream>>>(gbuf, c1, h1, (float*)nullptr, (__bf16*)nullptr, t);

    // Layer 2 mogrify. Step 0 reads x directly from h1 -> x2 (h1 must persist).
    for (int i = 0; i < 5; ++i) {
      const __bf16* Wb = m2Wb + (size_t)i * H_ * H_;
      const float*  bb = mog2_b + (size_t)i * H_;
      if (i == 0)
        k_mogrify<<<8, 256, 0, stream>>>(x2, h1, (long)H_, h2, Wb, bb);
      else if ((i & 1) == 0)
        k_mogrify<<<8, 256, 0, stream>>>(x2, x2, (long)H_, h2, Wb, bb);
      else
        k_mogrify<<<8, 256, 0, stream>>>(h2, h2, (long)H_, x2, Wb, bb);
    }
    k_gates<<<32, 256, 0, stream>>>(x2, h2, Wih2b, Whh2b, b_ih2, b_hh2, gbuf);
    k_cell<<<SN / 256, 256, 0, stream>>>(gbuf, c2, h2, hs_out, hsb, t);
  }

  // Final projection: (2048 x 1024) @ (1024 x 32000) + bias
  dim3 pg(V_ / 128, (B_ * T_) / 64);   // 250 x 32
  k_proj<<<pg, 256, 0, stream>>>(hsb, emb_b, fc_b, outF);
}
